// TransformerBlock_63239098466514
// MI455X (gfx1250) — matrix operand order repaired and checked by reading, not compiled, not run on hardware
//
#include <hip/hip_runtime.h>
#include <math.h>

#define BATCH 8
#define SEQ   2048
#define TCHN  512
#define NTOK  (BATCH*SEQ)
#define EPSLN 1e-5f

typedef __attribute__((ext_vector_type(16))) __bf16 v16bf;
typedef __attribute__((ext_vector_type(8)))  float  v8f;

union BFrag { uint4 u[2]; v16bf v; };

__device__ __forceinline__ unsigned short f2bf(float f) {
    union { float f; unsigned int u; } x; x.f = f;
    unsigned int r = x.u + 0x7FFFu + ((x.u >> 16) & 1u);
    return (unsigned short)(r >> 16);
}

// ---------------------------------------------------------------------------
// 1) q,k,v = x @ {wq,wk,wv}.T + bias   -> [NTOK,3] each
//    one wave per token; lane owns channels c = lane + 32*i (coalesced,
//    LDS-conflict-free); wave32 shfl reduction.
// ---------------------------------------------------------------------------
__global__ __launch_bounds__(256) void qkv_kernel(
    const float* __restrict__ x,
    const float* __restrict__ wq, const float* __restrict__ bq,
    const float* __restrict__ wk, const float* __restrict__ bk,
    const float* __restrict__ wv, const float* __restrict__ bv,
    float* __restrict__ q, float* __restrict__ k, float* __restrict__ v)
{
    __shared__ float sw[3 * 3 * TCHN];       // wq | wk | wv, each [3,512]
    for (int i = threadIdx.x; i < 3 * TCHN; i += 256) {
        sw[i]            = wq[i];
        sw[3*TCHN + i]   = wk[i];
        sw[6*TCHN + i]   = wv[i];
    }
    __syncthreads();

    const int wave = threadIdx.x >> 5, lane = threadIdx.x & 31;
    const int tok  = blockIdx.x * 8 + wave;
    const float* xr = x + (size_t)tok * TCHN;

    float acc[9];
#pragma unroll
    for (int i = 0; i < 9; ++i) acc[i] = 0.f;

#pragma unroll
    for (int i = 0; i < 16; ++i) {
        const int c = lane + 32 * i;
        const float xv = xr[c];
#pragma unroll
        for (int d = 0; d < 3; ++d) {
            acc[d]     += xv * sw[d*TCHN + c];
            acc[3 + d] += xv * sw[3*TCHN + d*TCHN + c];
            acc[6 + d] += xv * sw[6*TCHN + d*TCHN + c];
        }
    }
#pragma unroll
    for (int off = 16; off > 0; off >>= 1)
#pragma unroll
        for (int i = 0; i < 9; ++i) acc[i] += __shfl_xor(acc[i], off, 32);

    if (lane < 3) {
        q[tok*3 + lane] = acc[lane]     + bq[lane];
        k[tok*3 + lane] = acc[3 + lane] + bk[lane];
        v[tok*3 + lane] = acc[6 + lane] + bv[lane];
    }
}

// ---------------------------------------------------------------------------
// 2) wo_eff[c,d] = sum_h wo[c, h*3+d]   (heads concat folds onto wo)
// ---------------------------------------------------------------------------
__global__ __launch_bounds__(256) void fold1_kernel(
    const float* __restrict__ wo, float* __restrict__ wo_eff)
{
    const int c = blockIdx.x * 256 + threadIdx.x;   // 0..511
#pragma unroll
    for (int d = 0; d < 3; ++d) {
        float s = 0.f;
#pragma unroll
        for (int h = 0; h < 6; ++h) s += wo[c*18 + h*3 + d];
        wo_eff[c*3 + d] = s;
    }
}

// ---------------------------------------------------------------------------
// 3) weff = w1 @ wo_eff  [512,3] ;  beff = w1 @ bo + b1  [512]
// ---------------------------------------------------------------------------
__global__ __launch_bounds__(256) void fold2_kernel(
    const float* __restrict__ w1, const float* __restrict__ b1,
    const float* __restrict__ bo, const float* __restrict__ wo_eff,
    float* __restrict__ weff, float* __restrict__ beff)
{
    const int c = blockIdx.x * 256 + threadIdx.x;   // 0..511
    float a0 = 0.f, a1 = 0.f, a2 = 0.f, ab = 0.f;
    const float* wr = w1 + (size_t)c * TCHN;
    for (int j = 0; j < TCHN; ++j) {
        const float w = wr[j];
        a0 += w * wo_eff[j*3 + 0];
        a1 += w * wo_eff[j*3 + 1];
        a2 += w * wo_eff[j*3 + 2];
        ab += w * bo[j];
    }
    weff[c*3 + 0] = a0; weff[c*3 + 1] = a1; weff[c*3 + 2] = a2;
    beff[c] = ab + b1[c];
}

// ---------------------------------------------------------------------------
// 4) convert w2 [512,512] f32 -> bf16 (row-major preserved)
// ---------------------------------------------------------------------------
__global__ __launch_bounds__(256) void w2cvt_kernel(
    const float* __restrict__ w2, unsigned short* __restrict__ w2bf)
{
    const int i = blockIdx.x * 256 + threadIdx.x;
    w2bf[i] = f2bf(w2[i]);
}

// ---------------------------------------------------------------------------
// 5) attention: z = softmax(q k^T / sqrt(3)) v  -> [NTOK,3]
//    k,v of one batch staged in LDS (48KB); online softmax per thread.
// ---------------------------------------------------------------------------
__global__ __launch_bounds__(256) void attn_kernel(
    const float* __restrict__ q, const float* __restrict__ k,
    const float* __restrict__ v, float* __restrict__ z)
{
    __shared__ float sk[SEQ * 3];
    __shared__ float sv[SEQ * 3];
    const int b = blockIdx.x;
    const float* kb = k + (size_t)b * SEQ * 3;
    const float* vb = v + (size_t)b * SEQ * 3;
    for (int i = threadIdx.x; i < SEQ * 3; i += 256) { sk[i] = kb[i]; sv[i] = vb[i]; }
    __syncthreads();

    const int tok = b * SEQ + blockIdx.y * 256 + threadIdx.x;
    const float q0 = q[tok*3], q1 = q[tok*3 + 1], q2 = q[tok*3 + 2];
    const float scale = 0.5773502691896258f;   // 1/sqrt(3)

    float m = -INFINITY, l = 0.f, a0 = 0.f, a1 = 0.f, a2 = 0.f;
    for (int j = 0; j < SEQ; ++j) {
        const float k0 = sk[j*3], k1 = sk[j*3+1], k2 = sk[j*3+2];
        const float s = (q0*k0 + q1*k1 + q2*k2) * scale;
        if (s > m) {
            const float corr = __expf(m - s);       // 0 when m == -inf
            l  = l  * corr + 1.f;
            a0 = a0 * corr + sv[j*3];
            a1 = a1 * corr + sv[j*3+1];
            a2 = a2 * corr + sv[j*3+2];
            m = s;
        } else {
            const float p = __expf(s - m);
            l  += p;
            a0 += p * sv[j*3];
            a1 += p * sv[j*3+1];
            a2 += p * sv[j*3+2];
        }
    }
    const float inv = 1.f / l;
    z[tok*3 + 0] = a0 * inv;
    z[tok*3 + 1] = a1 * inv;
    z[tok*3 + 2] = a2 * inv;
}

// ---------------------------------------------------------------------------
// 6) r0 = x + z @ weff.T + beff ; r1 = LN(r0)  -> r0 (f32), r1 (bf16)
//    one wave per token, wave32 shfl mean/var reduction.
// ---------------------------------------------------------------------------
__global__ __launch_bounds__(256) void r0ln_kernel(
    const float* __restrict__ x, const float* __restrict__ z,
    const float* __restrict__ weff, const float* __restrict__ beff,
    const float* __restrict__ gamma, const float* __restrict__ beta,
    float* __restrict__ r0, unsigned short* __restrict__ r1bf)
{
    __shared__ float swe[TCHN * 3];
    __shared__ float sbe[TCHN], sg[TCHN], sb[TCHN];
    for (int i = threadIdx.x; i < TCHN * 3; i += 256) swe[i] = weff[i];
    for (int i = threadIdx.x; i < TCHN; i += 256) { sbe[i] = beff[i]; sg[i] = gamma[i]; sb[i] = beta[i]; }
    __syncthreads();

    const int wave = threadIdx.x >> 5, lane = threadIdx.x & 31;
    const int tok  = blockIdx.x * 8 + wave;
    const float z0 = z[tok*3], z1 = z[tok*3 + 1], z2 = z[tok*3 + 2];

    float r[16];
    float sum = 0.f, sq = 0.f;
#pragma unroll
    for (int i = 0; i < 16; ++i) {
        const int c = lane + 32 * i;
        const float val = x[(size_t)tok*TCHN + c]
                        + z0 * swe[c*3] + z1 * swe[c*3 + 1] + z2 * swe[c*3 + 2]
                        + sbe[c];
        r[i] = val; sum += val; sq += val * val;
    }
#pragma unroll
    for (int off = 16; off > 0; off >>= 1) {
        sum += __shfl_xor(sum, off, 32);
        sq  += __shfl_xor(sq,  off, 32);
    }
    const float mu   = sum * (1.f / TCHN);
    const float var  = sq * (1.f / TCHN) - mu * mu;
    const float rstd = rsqrtf(var + EPSLN);
#pragma unroll
    for (int i = 0; i < 16; ++i) {
        const int c = lane + 32 * i;
        r0[(size_t)tok*TCHN + c] = r[i];
        const float ln = (r[i] - mu) * rstd * sg[c] + sb[c];
        r1bf[(size_t)tok*TCHN + c] = f2bf(ln);
    }
}

// ---------------------------------------------------------------------------
// 7) r = r0 + r1 @ w2.T + b2 -> d_out   (WMMA bf16, f32 accumulate)
//    M=16384, N=512, K=512. Block = 8 waves, each wave one 16-row M tile,
//    32-col N strip staged in padded LDS; K loop of 16 x (2 WMMA).
// ---------------------------------------------------------------------------
#define NSTRIP 32
#define BROWP  (TCHN + 8)   // +8 ushort = 16B pad per row -> no LDS bank conflicts

__global__ __launch_bounds__(256) void gemm_kernel(
    const unsigned short* __restrict__ r1bf,   // [NTOK,512] bf16
    const unsigned short* __restrict__ w2bf,   // [512,512]  bf16 row-major
    const float* __restrict__ r0,
    const float* __restrict__ b2,
    float* __restrict__ rout)
{
    __shared__ __align__(16) unsigned short sB[NSTRIP * BROWP];
    const int n0 = blockIdx.y * NSTRIP;

    // cooperative load: 32 rows x 512 bf16, uint4 (8 ushort) granularity
    {
        const uint4* src = (const uint4*)(w2bf + (size_t)n0 * TCHN);
        for (int i = threadIdx.x; i < NSTRIP * (TCHN / 8); i += 256) {
            const int row = i >> 6;          // /64 chunks per row
            const int cch = i & 63;
            ((uint4*)(sB + row * BROWP))[cch] = src[(size_t)row * 64 + cch];
        }
    }
    __syncthreads();

    const int wave = threadIdx.x >> 5, lane = threadIdx.x & 31;
    const int mtile = blockIdx.x * 8 + wave;
    const int m0 = mtile * 16;

    const int arow  = m0 + (lane & 15);
    const int ahalf = lane >> 4;                       // 0: K 0-15, 1: K 16-31
    const int ncol  = lane & 15;
    const int khalf = lane >> 4;                       // B: 0: K 0-15, 1: K 16-31

    v8f c0 = {}; v8f c1 = {};
    const uint4* abase = (const uint4*)(r1bf + (size_t)arow * TCHN);

    for (int kt = 0; kt < 16; ++kt) {
        const int k0 = kt * 32;

        BFrag a;
        a.u[0] = abase[(k0 + ahalf * 16) >> 3];
        a.u[1] = abase[(k0 + ahalf * 16 + 8) >> 3];

        BFrag b0f, b1f;
        {
            const uint4* p0 = (const uint4*)(sB + ncol * BROWP + k0 + khalf * 16);
            b0f.u[0] = p0[0]; b0f.u[1] = p0[1];
            const uint4* p1 = (const uint4*)(sB + (16 + ncol) * BROWP + k0 + khalf * 16);
            b1f.u[0] = p1[0]; b1f.u[1] = p1[1];
        }

        c0 = __builtin_amdgcn_wmma_f32_16x16x32_bf16(false, a.v, false, b0f.v,
                                                     (short)0, c0, false, false);
        c1 = __builtin_amdgcn_wmma_f32_16x16x32_bf16(false, a.v, false, b1f.v,
                                                     (short)0, c1, false, false);
    }

    // epilogue: D layout -> n = lane%16, m = m0 + (lane/16)*8 + row
    const int nc   = lane & 15;
    const int mrow = m0 + (lane >> 4) * 8;
#pragma unroll
    for (int rr = 0; rr < 8; ++rr) {
        const int m = mrow + rr;
        {
            const int n = n0 + nc;
            const size_t idx = (size_t)m * TCHN + n;
            rout[idx] = c0[rr] + r0[idx] + b2[n];
        }
        {
            const int n = n0 + 16 + nc;
            const size_t idx = (size_t)m * TCHN + n;
            rout[idx] = c1[rr] + r0[idx] + b2[n];
        }
    }
}

// ---------------------------------------------------------------------------
// 8) out = LN(out) in place (row fully owned by one wave -> safe)
// ---------------------------------------------------------------------------
__global__ __launch_bounds__(256) void lnout_kernel(
    float* __restrict__ out, const float* __restrict__ gamma,
    const float* __restrict__ beta)
{
    const int wave = threadIdx.x >> 5, lane = threadIdx.x & 31;
    const int tok  = blockIdx.x * 8 + wave;
    float r[16];
    float sum = 0.f, sq = 0.f;
#pragma unroll
    for (int i = 0; i < 16; ++i) {
        const int c = lane + 32 * i;
        const float val = out[(size_t)tok*TCHN + c];
        r[i] = val; sum += val; sq += val * val;
    }
#pragma unroll
    for (int off = 16; off > 0; off >>= 1) {
        sum += __shfl_xor(sum, off, 32);
        sq  += __shfl_xor(sq,  off, 32);
    }
    const float mu   = sum * (1.f / TCHN);
    const float var  = sq * (1.f / TCHN) - mu * mu;
    const float rstd = rsqrtf(var + EPSLN);
#pragma unroll
    for (int i = 0; i < 16; ++i) {
        const int c = lane + 32 * i;
        out[(size_t)tok*TCHN + c] = (r[i] - mu) * rstd * gamma[c] + beta[c];
    }
}

// ---------------------------------------------------------------------------
extern "C" void kernel_launch(void* const* d_in, const int* in_sizes, int n_in,
                              void* d_out, int out_size, void* d_ws, size_t ws_size,
                              hipStream_t stream)
{
    const float* x     = (const float*)d_in[0];
    const float* wq    = (const float*)d_in[1];
    const float* bq    = (const float*)d_in[2];
    const float* wk    = (const float*)d_in[3];
    const float* bk    = (const float*)d_in[4];
    const float* wv    = (const float*)d_in[5];
    const float* bv    = (const float*)d_in[6];
    const float* wo    = (const float*)d_in[7];
    const float* bo    = (const float*)d_in[8];
    const float* w1    = (const float*)d_in[9];
    const float* b1    = (const float*)d_in[10];
    const float* w2    = (const float*)d_in[11];
    const float* b2    = (const float*)d_in[12];
    const float* gamma = (const float*)d_in[13];
    const float* beta  = (const float*)d_in[14];
    float* out = (float*)d_out;

    // workspace carve-up (256B aligned)
    char* ws = (char*)d_ws;
    size_t off = 0;
    auto alloc = [&](size_t bytes) -> char* {
        char* p = ws + off;
        off = (off + bytes + 255) & ~(size_t)255;
        return p;
    };
    float* q      = (float*)alloc((size_t)NTOK * 3 * 4);
    float* k      = (float*)alloc((size_t)NTOK * 3 * 4);
    float* v      = (float*)alloc((size_t)NTOK * 3 * 4);
    float* z      = (float*)alloc((size_t)NTOK * 3 * 4);
    float* wo_eff = (float*)alloc(TCHN * 3 * 4);
    float* weff   = (float*)alloc(TCHN * 3 * 4);
    float* beff   = (float*)alloc(TCHN * 4);
    float* r0     = (float*)alloc((size_t)NTOK * TCHN * 4);
    unsigned short* r1bf = (unsigned short*)alloc((size_t)NTOK * TCHN * 2);
    unsigned short* w2bf = (unsigned short*)alloc((size_t)TCHN * TCHN * 2);
    (void)ws_size; (void)in_sizes; (void)n_in; (void)out_size;

    qkv_kernel<<<NTOK / 8, 256, 0, stream>>>(x, wq, bq, wk, bk, wv, bv, q, k, v);
    fold1_kernel<<<2, 256, 0, stream>>>(wo, wo_eff);
    fold2_kernel<<<2, 256, 0, stream>>>(w1, b1, bo, wo_eff, weff, beff);
    w2cvt_kernel<<<(TCHN * TCHN) / 256, 256, 0, stream>>>(w2, w2bf);
    attn_kernel<<<dim3(BATCH, SEQ / 256), 256, 0, stream>>>(q, k, v, z);
    r0ln_kernel<<<NTOK / 8, 256, 0, stream>>>(x, z, weff, beff, gamma, beta, r0, r1bf);
    gemm_kernel<<<dim3(NTOK / 16 / 8, TCHN / NSTRIP), 256, 0, stream>>>(r1bf, w2bf, r0, b2, out);
    lnout_kernel<<<NTOK / 8, 256, 0, stream>>>(out, gamma, beta);
}
